// Attention_42004780155590
// MI455X (gfx1250) — compile-verified, matrix operand order checked by reading, not run on hardware
//
#include <hip/hip_runtime.h>

// GPT-2 attention block for MI455X (gfx1250, wave32, WMMA).
// Pipeline: [cvt f16] -> [QKV gemm, WMMA f16, async-LDS double buffer]
//           -> [flash attention, WMMA f16] -> [proj gemm, WMMA f16, async-LDS].
// Requires ws_size >= 48 MB (uses 50,331,648 bytes).

typedef __attribute__((ext_vector_type(16))) _Float16 v16h;
typedef __attribute__((ext_vector_type(8)))  _Float16 v8h;
typedef __attribute__((ext_vector_type(8)))  float    v8f;

#define WMMA_F16(a, b, c) \
  __builtin_amdgcn_wmma_f32_16x16x32_f16(false, (a), false, (b), (short)0, (c), false, false)

static __device__ __forceinline__ v16h cat8(v8h lo, v8h hi) {
  return __builtin_shufflevector(lo, hi, 0,1,2,3,4,5,6,7,8,9,10,11,12,13,14,15);
}
static __device__ __forceinline__ v8f zero8() {
  v8f z = {0.f,0.f,0.f,0.f,0.f,0.f,0.f,0.f};
  return z;
}

// CDNA5 async global->LDS copy, 16B per lane, tracked by ASYNCcnt (bypasses VGPRs).
// Generic LDS pointer low 32 bits == LDS byte offset (aperture bits live in [63:32]).
static __device__ __forceinline__ void async_cp16(const _Float16* g, _Float16* l) {
  asm volatile("global_load_async_to_lds_b128 %0, %1, off"
               :: "v"((unsigned)(unsigned long long)(size_t)l),
                  "v"((unsigned long long)(size_t)g)
               : "memory");
}
static __device__ __forceinline__ void wait_async0() {
  asm volatile("s_wait_asynccnt 0x0" ::: "memory");
}

// ---------------- problem constants ----------------
#define BB   2
#define SS   2048
#define EE   1024
#define HH   16
#define DH   64
#define NQKV 3072
#define MROWS (BB*SS)   // 4096

// ---------------- conversion kernels ----------------
__global__ void cvt_f16(const float* __restrict__ in, _Float16* __restrict__ out, int n) {
  int i = blockIdx.x * blockDim.x + threadIdx.x;
  if (i < n) out[i] = (_Float16)in[i];
}

// in: [K][N] row-major fp32 -> out: [N][K] f16 (pre-transposed weights)
__global__ void cvt_f16_T(const float* __restrict__ in, _Float16* __restrict__ out, int K, int N) {
  int i = blockIdx.x * blockDim.x + threadIdx.x;
  if (i < K * N) {
    int k = i / N, n = i % N;
    out[(size_t)n * K + k] = (_Float16)in[i];
  }
}

// Shared GEMM mainloop: C[128x128] = A[128xK] * Bt[128xK]^T, BK=32,
// double-buffered LDS filled by async DMA, 8 waves (2x4), 8 wmma/step.
#define GEMM_MAINLOOP(ACC)                                                        \
  const int tid = threadIdx.x;                                                    \
  const int wave = tid >> 5, lane = tid & 31, hl = lane >> 4, ln = lane & 15;     \
  const int wm = wave >> 2, wn = wave & 3;                                        \
  const int m0 = blockIdx.y * 128, n0 = blockIdx.x * 128;                         \
  const int r0 = tid >> 2, o0 = (tid & 3) << 3;                                   \
  const int r1 = (tid + 256) >> 2, o1 = ((tid + 256) & 3) << 3;                   \
  v8f ACC[4][2];                                                                  \
  _Pragma("unroll") for (int fa = 0; fa < 4; ++fa)                                \
    _Pragma("unroll") for (int fb = 0; fb < 2; ++fb) ACC[fa][fb] = zero8();       \
  /* prologue: async-fill buffer 0 */                                             \
  async_cp16(&A [(size_t)(m0 + r0) * EE + o0], &As[0][r0][o0]);                   \
  async_cp16(&Bt[(size_t)(n0 + r0) * EE + o0], &Bs[0][r0][o0]);                   \
  async_cp16(&A [(size_t)(m0 + r1) * EE + o1], &As[0][r1][o1]);                   \
  async_cp16(&Bt[(size_t)(n0 + r1) * EE + o1], &Bs[0][r1][o1]);                   \
  wait_async0();                                                                  \
  __syncthreads();                                                                \
  for (int k0 = 0; k0 < EE; k0 += 32) {                                           \
    const int buf = (k0 >> 5) & 1;                                                \
    const int kn = k0 + 32;                                                       \
    if (kn < EE) { /* issue next tile into other buffer while computing */        \
      async_cp16(&A [(size_t)(m0 + r0) * EE + kn + o0], &As[buf ^ 1][r0][o0]);    \
      async_cp16(&Bt[(size_t)(n0 + r0) * EE + kn + o0], &Bs[buf ^ 1][r0][o0]);    \
      async_cp16(&A [(size_t)(m0 + r1) * EE + kn + o1], &As[buf ^ 1][r1][o1]);    \
      async_cp16(&Bt[(size_t)(n0 + r1) * EE + kn + o1], &Bs[buf ^ 1][r1][o1]);    \
    }                                                                             \
    v16h af[4], bf[2];                                                            \
    _Pragma("unroll") for (int fm = 0; fm < 4; ++fm) {                            \
      const _Float16* p = &As[buf][wm * 64 + fm * 16 + ln][hl * 8];               \
      af[fm] = cat8(*(const v8h*)p, *(const v8h*)(p + 16));                       \
    }                                                                             \
    _Pragma("unroll") for (int fn = 0; fn < 2; ++fn) {                            \
      const _Float16* p = &Bs[buf][wn * 32 + fn * 16 + ln][hl * 8];               \
      bf[fn] = cat8(*(const v8h*)p, *(const v8h*)(p + 16));                       \
    }                                                                             \
    _Pragma("unroll") for (int fm = 0; fm < 4; ++fm)                              \
      _Pragma("unroll") for (int fn = 0; fn < 2; ++fn)                            \
        ACC[fm][fn] = WMMA_F16(af[fm], bf[fn], ACC[fm][fn]);                      \
    wait_async0();                                                                \
    __syncthreads();                                                              \
  }

// ---------------- QKV GEMM: [4096,1024] x [1024,3072] ----------------
__global__ __launch_bounds__(256)
void gemm_qkv(const _Float16* __restrict__ A, const _Float16* __restrict__ Bt,
              const float* __restrict__ bias,
              _Float16* __restrict__ q_h, _Float16* __restrict__ k_h,
              _Float16* __restrict__ vT, float* __restrict__ present) {
  __shared__ _Float16 As[2][128][48];   // 48-half stride: 96B rows, 16B-aligned chunks
  __shared__ _Float16 Bs[2][128][48];
  GEMM_MAINLOOP(acc)

  // epilogue: scatter q/k/vT (f16) + present (f32)
#pragma unroll
  for (int fm = 0; fm < 4; ++fm) {
    const int Rbase = m0 + wm * 64 + fm * 16 + 8 * hl;
#pragma unroll
    for (int fn = 0; fn < 2; ++fn) {
      const int Cg = n0 + wn * 32 + fn * 16 + ln;
      const int which = Cg >> 10, e = Cg & 1023, hh = e >> 6, d = e & 63;
      const float bs = bias[Cg];
#pragma unroll
      for (int jj = 0; jj < 8; ++jj) {
        const int r = Rbase + jj;
        const int b = r >> 11, s = r & 2047;
        const float val = acc[fm][fn][jj] + bs;
        const size_t hidx = (((size_t)(b * HH + hh)) * SS + s) * DH + d;
        if (which == 0) {
          q_h[hidx] = (_Float16)val;
        } else if (which == 1) {
          k_h[hidx] = (_Float16)val;
          present[hidx] = val;                             // present[0]
        } else {
          vT[(((size_t)(b * HH + hh)) * DH + d) * SS + s] = (_Float16)val;
          present[(size_t)BB * HH * SS * DH + hidx] = val; // present[1]
        }
      }
    }
  }
}

// ---------------- projection GEMM: [4096,1024] x [1024,1024] + bias ----------------
__global__ __launch_bounds__(256)
void gemm_proj(const _Float16* __restrict__ A, const _Float16* __restrict__ Bt,
               const float* __restrict__ bias, float* __restrict__ out) {
  __shared__ _Float16 As[2][128][48];
  __shared__ _Float16 Bs[2][128][48];
  GEMM_MAINLOOP(acc)

#pragma unroll
  for (int fm = 0; fm < 4; ++fm) {
    const int Rbase = m0 + wm * 64 + fm * 16 + 8 * hl;
#pragma unroll
    for (int fn = 0; fn < 2; ++fn) {
      const int Cg = n0 + wn * 32 + fn * 16 + ln;
      const float bs = bias[Cg];
#pragma unroll
      for (int jj = 0; jj < 8; ++jj)
        out[(size_t)(Rbase + jj) * EE + Cg] = acc[fm][fn][jj] + bs;
    }
  }
}

// ---------------- flash attention: one wave per 16-row q tile ----------------
__global__ __launch_bounds__(256)
void attn_kernel(const _Float16* __restrict__ q_h, const _Float16* __restrict__ k_h,
                 const _Float16* __restrict__ vT, _Float16* __restrict__ attn_h) {
  __shared__ _Float16 plds[8][16][72];   // per-wave P staging, 144B row stride
  const int tid = threadIdx.x;
  const int wl = tid >> 5, lane = tid & 31, hl = lane >> 4, ln = lane & 15;
  const int gw = blockIdx.x * 8 + wl;    // 4096 waves total
  const int qt = gw & 127;               // q tile within head
  const int bh = gw >> 7;                // (b,h) flat
  const int b = bh >> 4, hh = bh & 15;
  const _Float16* qp = q_h + (size_t)bh * SS * DH;
  const _Float16* kp = k_h + (size_t)bh * SS * DH;
  const _Float16* vp = vT + (size_t)bh * DH * SS;
  const int q0 = qt * 16;

  // q A-fragments (K-dim = Dh = 64 -> 2 steps of 32)
  v16h qa[2];
#pragma unroll
  for (int kk = 0; kk < 2; ++kk) {
    const _Float16* p = qp + (size_t)(q0 + ln) * DH + kk * 32 + hl * 8;
    qa[kk] = cat8(*(const v8h*)p, *(const v8h*)(p + 16));
  }

  v8f o[4];
#pragma unroll
  for (int fn = 0; fn < 4; ++fn) o[fn] = zero8();
  float mrow[8], lrow[8];
#pragma unroll
  for (int jj = 0; jj < 8; ++jj) { mrow[jj] = -3.0e38f; lrow[jj] = 0.f; }

  const int kv_end = q0 + 16;            // causal bound (exclusive)
  for (int kv0 = 0; kv0 < kv_end; kv0 += 64) {
    // scores: S = q @ K^T, 4 N-frags x 2 K-steps
    v8f sf[4];
#pragma unroll
    for (int fn = 0; fn < 4; ++fn) {
      v8f c = zero8();
#pragma unroll
      for (int kk = 0; kk < 2; ++kk) {
        const _Float16* p = kp + (size_t)(kv0 + fn * 16 + ln) * DH + kk * 32 + hl * 8;
        v16h kb = cat8(*(const v8h*)p, *(const v8h*)(p + 16));
        c = WMMA_F16(qa[kk], kb, c);
      }
      sf[fn] = c;
    }
    // scale 1/sqrt(64), causal mask, row max
    float tmax[8];
#pragma unroll
    for (int jj = 0; jj < 8; ++jj) tmax[jj] = -3.0e38f;
#pragma unroll
    for (int fn = 0; fn < 4; ++fn) {
      const int kcol = kv0 + fn * 16 + ln;
#pragma unroll
      for (int jj = 0; jj < 8; ++jj) {
        const int qrow = q0 + jj + 8 * hl;
        const float sv = (kcol <= qrow) ? sf[fn][jj] * 0.125f : -1.0e10f;
        sf[fn][jj] = sv;
        tmax[jj] = fmaxf(tmax[jj], sv);
      }
    }
#pragma unroll
    for (int off = 1; off < 16; off <<= 1)
#pragma unroll
      for (int jj = 0; jj < 8; ++jj)
        tmax[jj] = fmaxf(tmax[jj], __shfl_xor(tmax[jj], off, 32));

    // online softmax update
    float aexp[8], rsum[8];
#pragma unroll
    for (int jj = 0; jj < 8; ++jj) {
      const float mnew = fmaxf(mrow[jj], tmax[jj]);
      aexp[jj] = __expf(mrow[jj] - mnew);
      mrow[jj] = mnew;
      rsum[jj] = 0.f;
    }
#pragma unroll
    for (int fn = 0; fn < 4; ++fn)
#pragma unroll
      for (int jj = 0; jj < 8; ++jj) {
        const float pv = __expf(sf[fn][jj] - mrow[jj]);
        sf[fn][jj] = pv;
        rsum[jj] += pv;
      }
#pragma unroll
    for (int off = 1; off < 16; off <<= 1)
#pragma unroll
      for (int jj = 0; jj < 8; ++jj)
        rsum[jj] += __shfl_xor(rsum[jj], off, 32);
#pragma unroll
    for (int jj = 0; jj < 8; ++jj) lrow[jj] = lrow[jj] * aexp[jj] + rsum[jj];
#pragma unroll
    for (int fn = 0; fn < 4; ++fn)
#pragma unroll
      for (int jj = 0; jj < 8; ++jj) o[fn][jj] *= aexp[jj];

    // stage P (C-frag layout) -> LDS, re-read as A-fragments (wave-local only)
#pragma unroll
    for (int fn = 0; fn < 4; ++fn)
#pragma unroll
      for (int jj = 0; jj < 8; ++jj)
        plds[wl][jj + 8 * hl][fn * 16 + ln] = (_Float16)sf[fn][jj];
    asm volatile("s_wait_dscnt 0" ::: "memory");   // LDS in-order per wave; wait data landed

    // O += P @ V  (K-dim = 64 kv -> 2 steps; 4 d-frags)
#pragma unroll
    for (int kk = 0; kk < 2; ++kk) {
      const _Float16* pp = &plds[wl][ln][kk * 32 + hl * 8];
      v16h pa = cat8(*(const v8h*)pp, *(const v8h*)(pp + 16));
#pragma unroll
      for (int fn = 0; fn < 4; ++fn) {
        const _Float16* vpp = vp + (size_t)(fn * 16 + ln) * SS + kv0 + kk * 32 + hl * 8;
        v16h vb = cat8(*(const v8h*)vpp, *(const v8h*)(vpp + 16));
        o[fn] = WMMA_F16(pa, vb, o[fn]);
      }
    }
  }

  // normalize and write merged-head f16 activation [b*S+s][E]
#pragma unroll
  for (int fn = 0; fn < 4; ++fn)
#pragma unroll
    for (int jj = 0; jj < 8; ++jj) {
      const int s = q0 + jj + 8 * hl;
      const float val = o[fn][jj] / lrow[jj];
      attn_h[((size_t)(b * SS + s)) * EE + hh * DH + fn * 16 + ln] = (_Float16)val;
    }
}

// ---------------- host-side launcher ----------------
extern "C" void kernel_launch(void* const* d_in, const int* in_sizes, int n_in,
                              void* d_out, int out_size, void* d_ws, size_t ws_size,
                              hipStream_t stream) {
  (void)in_sizes; (void)n_in; (void)out_size; (void)ws_size;
  const float* x      = (const float*)d_in[0];
  const float* W_attn = (const float*)d_in[1];
  const float* b_attn = (const float*)d_in[2];
  const float* W_proj = (const float*)d_in[3];
  const float* b_proj = (const float*)d_in[4];

  float* out     = (float*)d_out;                       // [2,2048,1024] fp32
  float* present = out + (size_t)MROWS * EE;            // [2,2,16,2048,64] fp32

  // workspace carve (f16): 50,331,648 bytes total
  _Float16* ws     = (_Float16*)d_ws;
  _Float16* x_h    = ws;                                // 4096*1024
  _Float16* wa_t   = x_h + (size_t)MROWS * EE;          // 3072*1024 (transposed)
  _Float16* wp_t   = wa_t + (size_t)NQKV * EE;          // 1024*1024 (transposed)
  _Float16* q_h    = wp_t + (size_t)EE * EE;            // [b,h,s,d]
  _Float16* k_h    = q_h + (size_t)BB * HH * SS * DH;   // [b,h,s,d]
  _Float16* vT_h   = k_h + (size_t)BB * HH * SS * DH;   // [b,h,d,s]
  _Float16* attn_h = vT_h + (size_t)BB * HH * SS * DH;  // [b*s, e]

  const int n_x  = MROWS * EE;     // 4,194,304
  const int n_wa = EE * NQKV;      // 3,145,728
  const int n_wp = EE * EE;        // 1,048,576

  cvt_f16 <<<(n_x  + 255) / 256, 256, 0, stream>>>(x, x_h, n_x);
  cvt_f16_T<<<(n_wa + 255) / 256, 256, 0, stream>>>(W_attn, wa_t, EE, NQKV);
  cvt_f16_T<<<(n_wp + 255) / 256, 256, 0, stream>>>(W_proj, wp_t, EE, EE);

  gemm_qkv <<<dim3(NQKV / 128, MROWS / 128), 256, 0, stream>>>(
      x_h, wa_t, b_attn, q_h, k_h, vT_h, present);

  attn_kernel<<<(BB * HH * (SS / 16)) / 8, 256, 0, stream>>>(q_h, k_h, vT_h, attn_h);

  gemm_proj<<<dim3(EE / 128, MROWS / 128), 256, 0, stream>>>(attn_h, wp_t, b_proj, out);
}
